// lm_ot_56590489092495
// MI455X (gfx1250) — matrix operand order repaired and checked by reading, compiled, not run on hardware
//
#include <hip/hip_runtime.h>

// ---------------------------------------------------------------------------
// Problem dims (from reference setup)
// ---------------------------------------------------------------------------
#define BDIM   512          // batch
#define HDIM   1024         // hidden
#define NV     3000         // verbs
#define NVP    3008         // padded to 16 (188 tiles of 16)
#define SSL    32           // slots (nslots-1)
#define OUTD   512          // outputs feature dim
#define LAM    20.0f
#define STOPTHR 0.005f
#define MAXIT  1000
#define KSLICES 47          // phase-B k slices (47 * 64 = 3008)

typedef __bf16 bf16_t;
typedef __bf16 v16bf __attribute__((ext_vector_type(16)));
typedef float  v8f   __attribute__((ext_vector_type(8)));
typedef unsigned int uv4 __attribute__((ext_vector_type(4)));

// ---------------------------------------------------------------------------
// WMMA helpers (bf16 A/B, f32 accum, 16x16x32)
// ---------------------------------------------------------------------------
__device__ __forceinline__ v8f wmma_bf16(v16bf a, v16bf b, v8f c) {
  return __builtin_amdgcn_wmma_f32_16x16x32_bf16(
      /*neg_a=*/false, a, /*neg_b=*/false, b,
      /*c_mod=*/(short)0, c, /*reuse_a=*/false, /*reuse_b=*/false);
}

// A fragment: 16x32 tile from row-major [M][K] storage (two b128 loads/lane).
__device__ __forceinline__ v16bf load_fragA(const bf16_t* base, int stride,
                                            int row0, int k0, int lane) {
  int r = lane & 15, half = lane >> 4;
  const bf16_t* p = base + (size_t)(row0 + r) * stride + k0 + half * 16;
  union { uv4 q[2]; v16bf v; } u;
  u.q[0] = *(const uv4*)(p);
  u.q[1] = *(const uv4*)(p + 8);
  return u.v;
}

// B fragment: 32x16 tile from [N][K] storage (one 32B region/lane).
__device__ __forceinline__ v16bf load_fragB(const bf16_t* base, int stride,
                                            int col0, int k0, int lane) {
  int c = lane & 15, half = lane >> 4;
  const bf16_t* p = base + (size_t)(col0 + c) * stride + k0 + half * 16;
  union { uv4 q[2]; v16bf v; } u;
  u.q[0] = *(const uv4*)(p);
  u.q[1] = *(const uv4*)(p + 8);
  return u.v;
}

// 16x64 output tile GEMM with 2-stage software pipeline over K.
// Loads for step k+32 are issued before the WMMAs of step k so the
// scheduler can overlap L2 latency with matrix ops (partial loadcnt waits).
__device__ __forceinline__ void gemm16x64_bf16(const bf16_t* __restrict__ A, int lda,
                                               const bf16_t* __restrict__ B, int ldb,
                                               int row0, int col0, int K,
                                               int lane, v8f acc[4]) {
  v16bf a0 = load_fragA(A, lda, row0, 0, lane);
  v16bf b0[4];
#pragma unroll
  for (int q = 0; q < 4; ++q) b0[q] = load_fragB(B, ldb, col0 + q * 16, 0, lane);
  for (int kk = 0; kk < K; kk += 32) {
    int kn = (kk + 32 < K) ? kk + 32 : 0;   // last iter: redundant (discarded) load
    v16bf a1 = load_fragA(A, lda, row0, kn, lane);
    v16bf b1[4];
#pragma unroll
    for (int q = 0; q < 4; ++q) b1[q] = load_fragB(B, ldb, col0 + q * 16, kn, lane);
#pragma unroll
    for (int q = 0; q < 4; ++q) acc[q] = wmma_bf16(a0, b0[q], acc[q]);
    a0 = a1;
#pragma unroll
    for (int q = 0; q < 4; ++q) b0[q] = b1[q];
  }
}

// ---------------------------------------------------------------------------
// Block reductions (blockDim == 256)
// ---------------------------------------------------------------------------
__device__ __forceinline__ float block_reduce_sum(float v, float* red) {
  int t = threadIdx.x;
  red[t] = v; __syncthreads();
  for (int s = 128; s > 0; s >>= 1) { if (t < s) red[t] += red[t + s]; __syncthreads(); }
  float r = red[0]; __syncthreads();
  return r;
}
__device__ __forceinline__ float block_reduce_max(float v, float* red) {
  int t = threadIdx.x;
  red[t] = v; __syncthreads();
  for (int s = 128; s > 0; s >>= 1) { if (t < s) red[t] = fmaxf(red[t], red[t + s]); __syncthreads(); }
  float r = red[0]; __syncthreads();
  return r;
}

// ---------------------------------------------------------------------------
// Grid-wide barrier (persistent-kernel cooperative sync)
// ---------------------------------------------------------------------------
__device__ __forceinline__ void grid_sync(unsigned* cnt, unsigned* gen, unsigned nblocks) {
  __syncthreads();
  if (threadIdx.x == 0) {
    __threadfence();
    unsigned g = __hip_atomic_load(gen, __ATOMIC_RELAXED, __HIP_MEMORY_SCOPE_AGENT);
    unsigned prev = __hip_atomic_fetch_add(cnt, 1u, __ATOMIC_ACQ_REL, __HIP_MEMORY_SCOPE_AGENT);
    if (prev == nblocks - 1u) {
      __hip_atomic_store(cnt, 0u, __ATOMIC_RELAXED, __HIP_MEMORY_SCOPE_AGENT);
      __hip_atomic_fetch_add(gen, 1u, __ATOMIC_ACQ_REL, __HIP_MEMORY_SCOPE_AGENT);
    } else {
      while (__hip_atomic_load(gen, __ATOMIC_ACQUIRE, __HIP_MEMORY_SCOPE_AGENT) == g) {
        __builtin_amdgcn_s_sleep(2);
      }
    }
  }
  __syncthreads();
}

// ---------------------------------------------------------------------------
// K0: fp32 inputs -> bf16 (512 x 2048)
// ---------------------------------------------------------------------------
__global__ void cvt_inputs_kernel(const float* __restrict__ in, bf16_t* __restrict__ xb) {
  int i = blockIdx.x * 256 + threadIdx.x;   // grid covers 512*2048 exactly
  xb[i] = (bf16_t)in[i];
}

// ---------------------------------------------------------------------------
// K1: gather W_v = W_lm[verbs], bf16 convert, 1/(||row||+eps). Pad rows -> 0.
// ---------------------------------------------------------------------------
__global__ void gather_kernel(const float* __restrict__ W_lm, const int* __restrict__ verbs,
                              bf16_t* __restrict__ Wv, float* __restrict__ invn) {
  __shared__ float red[256];
  int i = blockIdx.x, tid = threadIdx.x;
  if (i < NV) {
    const float* src = W_lm + (size_t)verbs[i] * HDIM;
    float acc = 0.f;
    for (int k = tid; k < HDIM; k += 256) {
      float x = src[k];
      Wv[(size_t)i * HDIM + k] = (bf16_t)x;
      acc += x * x;
    }
    float n = block_reduce_sum(acc, red);
    if (tid == 0) invn[i] = 1.0f / (sqrtf(n) + 1e-8f);
  } else {
    for (int k = tid; k < HDIM; k += 256) Wv[(size_t)i * HDIM + k] = (bf16_t)0.0f;
    if (tid == 0) invn[i] = 0.f;
  }
}

// ---------------------------------------------------------------------------
// K2: topics -> normalized bf16 (32 x 1024)
// ---------------------------------------------------------------------------
__global__ void topics_kernel(const float* __restrict__ topics, bf16_t* __restrict__ tn) {
  __shared__ float red[256];
  int i = blockIdx.x, tid = threadIdx.x;
  const float* src = topics + (size_t)i * HDIM;
  float acc = 0.f;
  for (int k = tid; k < HDIM; k += 256) { float x = src[k]; acc += x * x; }
  float n = block_reduce_sum(acc, red);
  float inv = 1.0f / (sqrtf(n) + 1e-8f);
  for (int k = tid; k < HDIM; k += 256) tn[(size_t)i * HDIM + k] = (bf16_t)(src[k] * inv);
}

// ---------------------------------------------------------------------------
// K3 (WMMA): logits_h = X_h @ Wv^T for h in {0,1}. Each wave: 16x64 tile.
// ---------------------------------------------------------------------------
__global__ void logits_kernel(const bf16_t* __restrict__ Xb, const bf16_t* __restrict__ Wv,
                              float* __restrict__ l1, float* __restrict__ l2) {
  int wave = (blockIdx.x * 256 + threadIdx.x) >> 5;
  int lane = threadIdx.x & 31;
  if (wave >= 2 * 32 * 47) return;
  int h = wave / (32 * 47);
  int rem = wave % (32 * 47);
  int mt = rem / 47, nt4 = rem % 47;
  v8f acc[4] = {};
  gemm16x64_bf16(Xb + h * HDIM, 2 * HDIM, Wv, HDIM, mt * 16, nt4 * 64, HDIM, lane, acc);
  float* out = h ? l2 : l1;
  int n = lane & 15, mb = (lane >> 4) * 8;
  for (int q = 0; q < 4; ++q)
    for (int j = 0; j < 8; ++j)
      out[(size_t)(mt * 16 + mb + j) * NVP + nt4 * 64 + q * 16 + n] = acc[q][j];
}

// ---------------------------------------------------------------------------
// K4: per-row softmax of l1/2 (with /LM_TEMP), average -> bmatT (f32) + bmatB (bf16)
// ---------------------------------------------------------------------------
__global__ void softmax_o1_kernel(const float* __restrict__ l1, const float* __restrict__ l2,
                                  float* __restrict__ bmatT, bf16_t* __restrict__ bmatB) {
  __shared__ float red[256];
  int b = blockIdx.x, tid = threadIdx.x;
  const float* r1 = l1 + (size_t)b * NVP;
  const float* r2 = l2 + (size_t)b * NVP;
  float m1 = -1e30f, m2 = -1e30f;
  for (int v = tid; v < NV; v += 256) {
    m1 = fmaxf(m1, 0.5f * r1[v]);
    m2 = fmaxf(m2, 0.5f * r2[v]);
  }
  float M1 = block_reduce_max(m1, red);
  float M2 = block_reduce_max(m2, red);
  float s1 = 0.f, s2 = 0.f;
  for (int v = tid; v < NV; v += 256) {
    s1 += expf(0.5f * r1[v] - M1);
    s2 += expf(0.5f * r2[v] - M2);
  }
  float S1 = block_reduce_sum(s1, red);
  float S2 = block_reduce_sum(s2, red);
  float i1 = 0.5f / S1, i2 = 0.5f / S2;
  for (int v = tid; v < NVP; v += 256) {
    float val = 0.f;
    if (v < NV) val = expf(0.5f * r1[v] - M1) * i1 + expf(0.5f * r2[v] - M2) * i2;
    bmatT[(size_t)b * NVP + v] = val;
    bmatB[(size_t)b * NVP + v] = (bf16_t)val;
  }
}

// ---------------------------------------------------------------------------
// K5: o2 = softmax((outputs @ W_lin.T + b_lin)[:,1:33]); a_mat[s][b] = o2.T
// ---------------------------------------------------------------------------
__global__ void o2_kernel(const float* __restrict__ outputs, const float* __restrict__ W_lin,
                          const float* __restrict__ b_lin, float* __restrict__ a_mat) {
  int wave = (blockIdx.x * 256 + threadIdx.x) >> 5;
  int lane = threadIdx.x & 31;
  if (wave >= BDIM) return;
  int b = wave, s = lane + 1;               // slots 1..32
  const float* orow = outputs + (size_t)b * OUTD;
  const float* wrow = W_lin + (size_t)s * OUTD;
  float acc = 0.f;
  for (int k = 0; k < OUTD; ++k) acc += orow[k] * wrow[k];
  acc += b_lin[s];
  float m = acc;
  for (int off = 16; off; off >>= 1) m = fmaxf(m, __shfl_xor(m, off, 32));
  float e = expf(acc - m);
  float sum = e;
  for (int off = 16; off; off >>= 1) sum += __shfl_xor(sum, off, 32);
  a_mat[(size_t)lane * BDIM + b] = e / sum;
}

// ---------------------------------------------------------------------------
// K6 (WMMA): Mmat[s][v] = 1 - (tn_s . Wv_v) * invn[v]   (32 x 3008)
// ---------------------------------------------------------------------------
__global__ void m_kernel(const bf16_t* __restrict__ tn, const bf16_t* __restrict__ Wv,
                         const float* __restrict__ invn, float* __restrict__ Mmat) {
  int wave = (blockIdx.x * 256 + threadIdx.x) >> 5;
  int lane = threadIdx.x & 31;
  if (wave >= 2 * 47) return;
  int mt = wave / 47, nt4 = wave % 47;
  v8f acc[4] = {};
  gemm16x64_bf16(tn, HDIM, Wv, HDIM, mt * 16, nt4 * 64, HDIM, lane, acc);
  int n = lane & 15, mb = (lane >> 4) * 8;
  for (int q = 0; q < 4; ++q) {
    int v = nt4 * 64 + q * 16 + n;
    float iv = invn[v];
    for (int j = 0; j < 8; ++j)
      Mmat[(size_t)(mt * 16 + mb + j) * NVP + v] = 1.0f - acc[q][j] * iv;
  }
}

// ---------------------------------------------------------------------------
// K7: K = exp(-lam*M) (bf16 + transposed copy), KM = K*M (f32). Pad -> 0.
// ---------------------------------------------------------------------------
__global__ void expk_kernel(const float* __restrict__ Mmat, bf16_t* __restrict__ Kbf,
                            bf16_t* __restrict__ Ktbf, float* __restrict__ KM) {
  int idx = blockIdx.x * 256 + threadIdx.x;          // covers 32*3008 exactly
  int s = idx / NVP, v = idx % NVP;
  float Mv = Mmat[idx];
  float Kv = (v < NV) ? expf(-LAM * Mv) : 0.f;
  Kbf[idx] = (bf16_t)Kv;
  Ktbf[(size_t)v * SSL + s] = (bf16_t)Kv;
  KM[idx] = (v < NV) ? Kv * Mv : 0.f;
}

// ---------------------------------------------------------------------------
// K8: init u = 1/32 (f32 + transposed bf16) and control words
// ---------------------------------------------------------------------------
__global__ void init_kernel(float* __restrict__ u, bf16_t* __restrict__ ubt,
                            unsigned* __restrict__ ctrl) {
  int i = blockIdx.x * 256 + threadIdx.x;
  if (i < SSL * BDIM) {
    u[i] = 0.03125f;
    ubt[i] = (bf16_t)0.03125f;
  }
  if (i == 0) {
    ctrl[0] = 0u;                       // barrier arrive count
    ctrl[1] = 0u;                       // barrier generation
    ctrl[2] = 0u;                       // (spare)
    ctrl[3] = __float_as_uint(1.0f);    // err
  }
}

// ---------------------------------------------------------------------------
// K9: persistent Sinkhorn. 94 blocks x 256 threads = 752 waves.
// Phase A  (WMMA, k=32, 6016 tiles): r[b][v] = bmat[b][v] / (K^T u)[b][v]
// Phase B1 (WMMA, split-k, 3008 tasks): per-slice partial y
// Phase B2: u = a / sum_slices(y)   (fixed-order, deterministic)
// Check every 20 iters per reference schedule; deterministic final reduce.
// ---------------------------------------------------------------------------
__global__ void __launch_bounds__(256)
sinkhorn_kernel(const bf16_t* __restrict__ Ktbf, const bf16_t* __restrict__ Kbf,
                const float* __restrict__ KM, const float* __restrict__ bmatT,
                const bf16_t* __restrict__ bmatB, const float* __restrict__ amat,
                float* __restrict__ u, bf16_t* __restrict__ ubt,
                bf16_t* __restrict__ rT, float* __restrict__ vT,
                float* __restrict__ wbuf, float* __restrict__ ybuf,
                unsigned* __restrict__ ctrl, float* __restrict__ out) {
  const int tid = threadIdx.x;
  const int wave = (blockIdx.x * blockDim.x + tid) >> 5;
  const int lane = tid & 31;
  const int nwaves = (gridDim.x * blockDim.x) >> 5;
  const unsigned nblocks = gridDim.x;
  const int n16 = lane & 15, mb8 = (lane >> 4) << 3;
  const int NT = (BDIM / 16) * (NVP / 16);   // 6016 phase-A tiles
  const int NB1 = 64 * KSLICES;              // 3008 phase-B1 tasks

  float err = 1.0f;
  for (int cpt = 1; cpt <= MAXIT; ++cpt) {
    const bool check = ((cpt % 20) == 1) || (cpt == MAXIT);
    const int reps = check ? 2 : 1;
    for (int rep = 0; rep < reps; ++rep) {
      const bool storeV = check && (rep == 1);
      // ---- phase A: r = b / (K^T u)
      for (int t = wave; t < NT; t += nwaves) {
        int mt = t / (NVP / 16), nt = t % (NVP / 16);
        v16bf a = load_fragA(ubt, SSL, mt * 16, 0, lane);
        v16bf b = load_fragB(Ktbf, SSL, nt * 16, 0, lane);
        int v = nt * 16 + n16;
        bool valid = v < NV;
        float num[8];                       // numerators prefetched before WMMA
#pragma unroll
        for (int j = 0; j < 8; ++j)
          num[j] = valid ? (float)bmatB[(size_t)(mt * 16 + mb8 + j) * NVP + v] : 0.f;
        v8f c = {};
        c = wmma_bf16(a, b, c);
#pragma unroll
        for (int j = 0; j < 8; ++j) {
          int bb = mt * 16 + mb8 + j;
          float r = valid ? num[j] / c[j] : 0.0f;
          rT[(size_t)bb * NVP + v] = (bf16_t)r;
          if (storeV) vT[(size_t)bb * NVP + v] = r;
        }
      }
      grid_sync(ctrl, ctrl + 1, nblocks);
      // ---- phase B1: per-slice partial y[s][b] (split-k to kill latency)
      for (int t = wave; t < NB1; t += nwaves) {
        int tile = t / KSLICES, sl = t % KSLICES;
        int mt = tile >> 5, nt = tile & 31;
        int kk = sl * 64;
        v16bf a0 = load_fragA(Kbf, NVP, mt * 16, kk, lane);
        v16bf b0 = load_fragB(rT, NVP, nt * 16, kk, lane);
        v16bf a1 = load_fragA(Kbf, NVP, mt * 16, kk + 32, lane);
        v16bf b1 = load_fragB(rT, NVP, nt * 16, kk + 32, lane);
        v8f c = {};
        c = wmma_bf16(a0, b0, c);
        c = wmma_bf16(a1, b1, c);
        int bb = nt * 16 + n16;
#pragma unroll
        for (int j = 0; j < 8; ++j)
          ybuf[(size_t)sl * (SSL * BDIM) + (size_t)(mt * 16 + mb8 + j) * BDIM + bb] = c[j];
      }
      grid_sync(ctrl, ctrl + 1, nblocks);
      // ---- phase B2: u = a / sum_slices(y)  (one thread per element)
      {
        int i = blockIdx.x * 256 + tid;
        if (i < SSL * BDIM) {
          float acc = 0.f;
          for (int sl = 0; sl < KSLICES; ++sl) acc += ybuf[(size_t)sl * (SSL * BDIM) + i];
          int s = i >> 9, bb = i & (BDIM - 1);
          float uu = amat[i] / acc;
          u[i] = uu;
          ubt[(size_t)bb * SSL + s] = (bf16_t)uu;
        }
      }
      grid_sync(ctrl, ctrl + 1, nblocks);
    }
    if (check) {
      // ---- w3 = K^T u (f32, into wbuf); also reset err accumulator
      if (blockIdx.x == 0 && tid == 0)
        __hip_atomic_store(ctrl + 3, 0u, __ATOMIC_RELAXED, __HIP_MEMORY_SCOPE_AGENT);
      for (int t = wave; t < NT; t += nwaves) {
        int mt = t / (NVP / 16), nt = t % (NVP / 16);
        v16bf a = load_fragA(ubt, SSL, mt * 16, 0, lane);
        v16bf b = load_fragB(Ktbf, SSL, nt * 16, 0, lane);
        v8f c = {};
        c = wmma_bf16(a, b, c);
        int v = nt * 16 + n16;
#pragma unroll
        for (int j = 0; j < 8; ++j)
          wbuf[(size_t)(mt * 16 + mb8 + j) * NVP + v] = c[j];
      }
      grid_sync(ctrl, ctrl + 1, nblocks);
      // ---- err = max_b sum_v |v*w3 - b|
      for (int b = wave; b < BDIM; b += nwaves) {
        const float* vr = vT + (size_t)b * NVP;
        const float* wr = wbuf + (size_t)b * NVP;
        const float* br = bmatT + (size_t)b * NVP;
        float s = 0.f;
        for (int v = lane; v < NV; v += 32) s += fabsf(vr[v] * wr[v] - br[v]);
        for (int off = 16; off; off >>= 1) s += __shfl_xor(s, off, 32);
        if (lane == 0) atomicMax(ctrl + 3, __float_as_uint(s));
      }
      grid_sync(ctrl, ctrl + 1, nblocks);
      err = __uint_as_float(__hip_atomic_load(ctrl + 3, __ATOMIC_RELAXED,
                                              __HIP_MEMORY_SCOPE_AGENT));
      if (!(err > STOPTHR)) break;
    }
  }

  // ---- final: zu[s,b] = u[s][b] * sum_v KM[s][v]*v[b][v]  (single writer each)
  for (int t = wave; t < SSL * BDIM; t += nwaves) {
    int s = t >> 9, b = t & (BDIM - 1);
    const float* kmr = KM + (size_t)s * NVP;
    const float* vr = vT + (size_t)b * NVP;
    float acc = 0.f;
    for (int v = lane; v < NV; v += 32) acc += kmr[v] * vr[v];
    for (int off = 16; off; off >>= 1) acc += __shfl_xor(acc, off, 32);
    if (lane == 0) wbuf[t] = u[(size_t)s * BDIM + b] * acc;
  }
  grid_sync(ctrl, ctrl + 1, nblocks);
  if (blockIdx.x == 0) {
    __shared__ float red[256];
    float s = 0.f;
    for (int i = tid; i < SSL * BDIM; i += 256) s += wbuf[i];  // fixed order
    float tot = block_reduce_sum(s, red);
    if (tid == 0) out[0] = tot / (float)BDIM;
  }
}

// ---------------------------------------------------------------------------
// Host launcher
// ---------------------------------------------------------------------------
extern "C" void kernel_launch(void* const* d_in, const int* in_sizes, int n_in,
                              void* d_out, int out_size, void* d_ws, size_t ws_size,
                              hipStream_t stream) {
  (void)in_sizes; (void)n_in; (void)out_size; (void)ws_size;
  const float* inputs  = (const float*)d_in[0];
  const float* outputs = (const float*)d_in[1];
  const float* W_lm    = (const float*)d_in[2];
  const float* W_lin   = (const float*)d_in[3];
  const float* b_lin   = (const float*)d_in[4];
  const float* topics  = (const float*)d_in[5];
  const int*   verbs   = (const int*)d_in[6];
  float* out = (float*)d_out;

  char* w = (char*)d_ws;
  auto carve = [&](size_t bytes) -> char* {
    char* p = w;
    w += (bytes + 255) & ~(size_t)255;
    return p;
  };
  bf16_t* Xb    = (bf16_t*)carve((size_t)BDIM * 2 * HDIM * 2);
  bf16_t* Wv    = (bf16_t*)carve((size_t)NVP * HDIM * 2);
  float*  invn  = (float*) carve((size_t)NVP * 4);
  bf16_t* tn    = (bf16_t*)carve((size_t)SSL * HDIM * 2);
  float*  l1    = (float*) carve((size_t)BDIM * NVP * 4);
  float*  l2    = (float*) carve((size_t)BDIM * NVP * 4);
  float*  bmatT = (float*) carve((size_t)BDIM * NVP * 4);
  bf16_t* bmatB = (bf16_t*)carve((size_t)BDIM * NVP * 2);
  float*  amat  = (float*) carve((size_t)SSL * BDIM * 4);
  float*  Mmat  = (float*) carve((size_t)SSL * NVP * 4);
  bf16_t* Kbf   = (bf16_t*)carve((size_t)SSL * NVP * 2);
  bf16_t* Ktbf  = (bf16_t*)carve((size_t)NVP * SSL * 2);
  float*  KM    = (float*) carve((size_t)SSL * NVP * 4);
  bf16_t* rT    = (bf16_t*)carve((size_t)BDIM * NVP * 2);
  float*  vT    = (float*) carve((size_t)BDIM * NVP * 4);
  float*  wbuf  = (float*) carve((size_t)BDIM * NVP * 4);
  float*  ybuf  = (float*) carve((size_t)KSLICES * SSL * BDIM * 4);
  float*  u     = (float*) carve((size_t)SSL * BDIM * 4);
  bf16_t* ubt   = (bf16_t*)carve((size_t)BDIM * SSL * 2);
  unsigned* ctrl = (unsigned*)carve(256);

  cvt_inputs_kernel<<<(BDIM * 2 * HDIM) / 256, 256, 0, stream>>>(inputs, Xb);
  gather_kernel<<<NVP, 256, 0, stream>>>(W_lm, verbs, Wv, invn);
  topics_kernel<<<SSL, 256, 0, stream>>>(topics, tn);
  logits_kernel<<<(2 * 32 * 47) / 8, 256, 0, stream>>>(Xb, Wv, l1, l2);
  softmax_o1_kernel<<<BDIM, 256, 0, stream>>>(l1, l2, bmatT, bmatB);
  o2_kernel<<<(BDIM * 32) / 256, 256, 0, stream>>>(outputs, W_lin, b_lin, amat);
  m_kernel<<<12, 256, 0, stream>>>(tn, Wv, invn, Mmat);
  expk_kernel<<<(SSL * NVP) / 256, 256, 0, stream>>>(Mmat, Kbf, Ktbf, KM);
  init_kernel<<<64, 256, 0, stream>>>(u, ubt, ctrl);
  sinkhorn_kernel<<<94, 256, 0, stream>>>(Ktbf, Kbf, KM, bmatT, bmatB, amat, u, ubt,
                                          rT, vT, wbuf, ybuf, ctrl, out);
}
